// AttentionFlow_29832842838018
// MI455X (gfx1250) — matrix operand order repaired and checked by reading, not compiled, not run on hardware
//
#include <hip/hip_runtime.h>
#include <hip/hip_bf16.h>

typedef __attribute__((ext_vector_type(16))) __bf16 v16bf;
typedef __attribute__((ext_vector_type(8)))  float  v8f;

#define BB 32
#define LL 1024
#define QQ 128
#define HH 512
#define H4 2048
#define NEG_INF_F (-1e30f)

__device__ __forceinline__ float lane_bcast(float v, int src) {
  return __shfl(v, src, 32);
}

// ---------------------------------------------------------------------------
// Kernel 1: scores (bf16 WMMA, K=512) -> softmax over q -> P@Q (bf16 WMMA)
// writes out segments 1 (attended_question) and 2 (context*attended_question)
// plus per-row max logits for the q2c path.
// grid = B * 8 blocks of 256 threads (8 waves); each wave: 16 rows of L.
// Question is converted to bf16 into LDS once per block. Context A-fragments
// are built once, in two K-halves of 8 chunks each to stay under the VGPR
// spill threshold (aC 64 + sc 64 + pipeline temps < 256).
// ---------------------------------------------------------------------------
__global__ __launch_bounds__(256) void bidaf_scores_kernel(
    const float* __restrict__ context, const float* __restrict__ question,
    const unsigned char* __restrict__ qmask, const float* __restrict__ w_sim,
    float* __restrict__ out, float* __restrict__ q2c_logits)
{
  __shared__ __bf16 s_q[QQ * HH];      // question[b] as bf16, row-major (128 KB)
  __shared__ __bf16 s_P[8][16 * QQ];   // per-wave 16x128 probability tiles (32 KB)
  __shared__ float  s_part[256];
  __shared__ float  s_qwq[QQ];

  const int b    = blockIdx.x >> 3;
  const int lblk = blockIdx.x & 7;
  const int wave = threadIdx.x >> 5;
  const int lane = threadIdx.x & 31;
  const int half = lane >> 4;          // 0: lanes 0-15, 1: lanes 16-31
  const int mlo  = lane & 15;
  const int lbase = lblk * 128 + wave * 16;

  const float* wc = w_sim;
  const float* wq = w_sim + HH;
  const float* we = w_sim + 2 * HH;

  // ---- Phase A: question -> bf16 LDS, and qwq[q] = question[b,q,:].wq ----
  {
    int q  = threadIdx.x >> 1;           // 0..127
    int hc = (threadIdx.x & 1) * 256;    // half-row
    const float* qp = question + ((size_t)b * QQ + q) * HH + hc;
    const float* wp = wq + hc;
    __bf16* dst = &s_q[q * HH + hc];
    float s = 0.f;
    #pragma unroll 8
    for (int h = 0; h < 256; ++h) {
      float v = qp[h];
      s += v * wp[h];
      dst[h] = (__bf16)v;
    }
    s_part[threadIdx.x] = s;
  }
  __syncthreads();
  if (threadIdx.x < QQ) {
    float s = s_part[2 * threadIdx.x] + s_part[2 * threadIdx.x + 1];
    s_qwq[threadIdx.x] = s + (qmask[b * QQ + threadIdx.x] ? 0.f : NEG_INF_F);
  }
  __syncthreads();

  // this lane supplies A-matrix row m = mlo (context row lbase+mlo)
  const float* crow = context + ((size_t)b * LL + lbase + mlo) * HH;

  // ---- Phase B/C fused: two K-halves; build 8 A-chunks, then 8x8 WMMAs ----
  v8f sc[8];
  #pragma unroll
  for (int t = 0; t < 8; ++t) sc[t] = (v8f){};
  float cwc = 0.f;

  #pragma unroll 1
  for (int kh = 0; kh < 2; ++kh) {
    const int kho = 256 * kh;
    // A fragments for this K-half (context .* we), plus cwc partial
    v16bf aC[8];
    #pragma unroll
    for (int kc = 0; kc < 8; ++kc) {
      int r0 = kho + 32 * kc + 16 * half;
      int r1 = r0 + 8;
      #pragma unroll
      for (int j = 0; j < 8; ++j) {
        float c0 = crow[r0 + j];
        float c1 = crow[r1 + j];
        cwc += c0 * wc[r0 + j] + c1 * wc[r1 + j];
        aC[kc][j]     = (__bf16)(c0 * we[r0 + j]);
        aC[kc][8 + j] = (__bf16)(c1 * we[r1 + j]);
      }
    }
    // accumulate this K-half into all 8 q-tiles
    #pragma unroll
    for (int t = 0; t < 8; ++t) {
      const __bf16* qrow = &s_q[(16 * t + mlo) * HH + 16 * half + kho];
      v8f acc = sc[t];
      #pragma unroll
      for (int kc = 0; kc < 8; ++kc) {
        v16bf bq = *(const v16bf*)(qrow + 32 * kc);   // 32B aligned LDS load
        acc = __builtin_amdgcn_wmma_f32_16x16x32_bf16(
            false, aC[kc], false, bq, (short)0, acc, false, false);
      }
      sc[t] = acc;
    }
  }
  cwc += __shfl_xor(cwc, 16, 32);      // combine the two lane-halves of K

  // rank-1 terms: + cwc[m] + qwq[n]
  #pragma unroll
  for (int t = 0; t < 8; ++t) {
    float qn = s_qwq[mlo + 16 * t];
    #pragma unroll
    for (int v = 0; v < 8; ++v)
      sc[t][v] += lane_bcast(cwc, v + 8 * half) + qn;
  }

  // ---- Phase D: softmax over q (n across 16 lanes x 8 tiles) ----
  float rmax[8];
  #pragma unroll
  for (int v = 0; v < 8; ++v) {
    float m = sc[0][v];
    #pragma unroll
    for (int t = 1; t < 8; ++t) m = fmaxf(m, sc[t][v]);
    #pragma unroll
    for (int o = 1; o < 16; o <<= 1) m = fmaxf(m, __shfl_xor(m, o, 32));
    rmax[v] = m;
  }
  if (mlo == 0) {
    #pragma unroll
    for (int v = 0; v < 8; ++v)
      q2c_logits[b * LL + lbase + v + 8 * half] = rmax[v];
  }
  float rinv[8];
  #pragma unroll
  for (int v = 0; v < 8; ++v) {
    float s = 0.f;
    #pragma unroll
    for (int t = 0; t < 8; ++t) {
      float e = __expf(sc[t][v] - rmax[v]);
      sc[t][v] = e;
      s += e;
    }
    #pragma unroll
    for (int o = 1; o < 16; o <<= 1) s += __shfl_xor(s, o, 32);
    rinv[v] = 1.f / s;
  }

  // ---- stash P (16x128) to per-wave LDS as bf16 (D-layout -> row major) ----
  {
    __bf16* pw = &s_P[wave][0];
    #pragma unroll
    for (int t = 0; t < 8; ++t)
      #pragma unroll
      for (int v = 0; v < 8; ++v)
        pw[(v + 8 * half) * QQ + mlo + 16 * t] = (__bf16)(sc[t][v] * rinv[v]);
  }

  // ---- load P back as A fragments (4 K-chunks of 32) ----
  v16bf aP[4];
  {
    const __bf16* pm = &s_P[wave][mlo * QQ];
    #pragma unroll
    for (int kq = 0; kq < 4; ++kq) {
      int r0 = 32 * kq + 16 * half;
      #pragma unroll
      for (int j = 0; j < 8; ++j) {
        aP[kq][j]     = pm[r0 + j];
        aP[kq][8 + j] = pm[r0 + 8 + j];
      }
    }
  }

  // ---- Phase E: AQ(16x512) = P(16x128) @ Q(128x512), B-frags from LDS ----
  const size_t outrow0 = ((size_t)b * LL + lbase) * H4;
  #pragma unroll 1
  for (int t2 = 0; t2 < 32; ++t2) {
    v8f acc = {};
    #pragma unroll
    for (int kq = 0; kq < 4; ++kq) {
      v16bf bq;
      const __bf16* qp = &s_q[(32 * kq + 16 * half) * HH + 16 * t2 + mlo];
      #pragma unroll
      for (int i = 0; i < 16; ++i) bq[i] = qp[i * HH];
      acc = __builtin_amdgcn_wmma_f32_16x16x32_bf16(
          false, aP[kq], false, bq, (short)0, acc, false, false);
    }
    int h = 16 * t2 + mlo;
    #pragma unroll
    for (int v = 0; v < 8; ++v) {
      int m = v + 8 * half;
      float ctx = context[((size_t)b * LL + lbase + m) * HH + h];
      float aq  = acc[v];
      size_t row = outrow0 + (size_t)m * H4;
      out[row + HH + h]     = aq;        // segment 1: attended_question
      out[row + 2 * HH + h] = ctx * aq;  // segment 2: context * attended_question
    }
  }
}

// ---------------------------------------------------------------------------
// Kernel 2: per-batch q2c softmax over L, then ac[b,h] = sum_l w_l * C[b,l,h]
// ---------------------------------------------------------------------------
__global__ __launch_bounds__(256) void bidaf_q2c_kernel(
    const float* __restrict__ context, const unsigned char* __restrict__ cmask,
    const float* __restrict__ logits, float* __restrict__ ac)
{
  __shared__ float s_w[LL];
  __shared__ float s_red[256];
  const int b = blockIdx.x, tid = threadIdx.x;

  float loc[4];
  float mx = NEG_INF_F;
  #pragma unroll
  for (int i = 0; i < 4; ++i) {
    int l = tid * 4 + i;
    float v = logits[b * LL + l];
    v = cmask[b * LL + l] ? v : NEG_INF_F;
    loc[i] = v;
    mx = fmaxf(mx, v);
  }
  s_red[tid] = mx;
  __syncthreads();
  for (int o = 128; o > 0; o >>= 1) {
    if (tid < o) s_red[tid] = fmaxf(s_red[tid], s_red[tid + o]);
    __syncthreads();
  }
  mx = s_red[0];
  __syncthreads();

  float se = 0.f;
  #pragma unroll
  for (int i = 0; i < 4; ++i) {
    float e = __expf(loc[i] - mx);
    s_w[tid * 4 + i] = e;
    se += e;
  }
  s_red[tid] = se;
  __syncthreads();
  for (int o = 128; o > 0; o >>= 1) {
    if (tid < o) s_red[tid] += s_red[tid + o];
    __syncthreads();
  }
  const float inv = 1.f / s_red[0];
  __syncthreads();

  #pragma unroll
  for (int c = 0; c < 2; ++c) {
    int h = tid + 256 * c;
    float s = 0.f;
    for (int l = 0; l < LL; ++l)
      s += s_w[l] * context[((size_t)b * LL + l) * HH + h];
    ac[b * HH + h] = s * inv;
  }
}

// ---------------------------------------------------------------------------
// Kernel 3: out segment 0 = context, segment 3 = context * ac[b,:]
// ---------------------------------------------------------------------------
__global__ __launch_bounds__(256) void bidaf_combine_kernel(
    const float* __restrict__ context, const float* __restrict__ ac,
    float* __restrict__ out)
{
  size_t idx = (size_t)blockIdx.x * 256 + threadIdx.x;   // float4 index
  size_t e = idx * 4;
  int b = (int)(e / ((size_t)LL * HH));
  size_t r = e % ((size_t)LL * HH);
  int l = (int)(r / HH);
  int h = (int)(r % HH);

  float4 c = ((const float4*)context)[idx];
  float4 a = ((const float4*)ac)[(b * HH + h) >> 2];
  size_t base = ((size_t)b * LL + l) * H4;
  *(float4*)(out + base + h) = c;
  *(float4*)(out + base + 3 * HH + h) =
      make_float4(c.x * a.x, c.y * a.y, c.z * a.z, c.w * a.w);
}

// ---------------------------------------------------------------------------
extern "C" void kernel_launch(void* const* d_in, const int* in_sizes, int n_in,
                              void* d_out, int out_size, void* d_ws, size_t ws_size,
                              hipStream_t stream)
{
  const float* context          = (const float*)d_in[0];
  const float* question         = (const float*)d_in[1];
  const unsigned char* cmask    = (const unsigned char*)d_in[2];
  const unsigned char* qmask    = (const unsigned char*)d_in[3];
  const float* w_sim            = (const float*)d_in[4];
  float* out = (float*)d_out;

  float* q2c_logits = (float*)d_ws;            // B*L floats
  float* ac         = q2c_logits + BB * LL;    // B*H floats

  bidaf_scores_kernel<<<BB * 8, 256, 0, stream>>>(context, question, qmask,
                                                  w_sim, out, q2c_logits);
  bidaf_q2c_kernel<<<BB, 256, 0, stream>>>(context, cmask, q2c_logits, ac);

  const int n4 = (BB * LL * HH) / 4;
  bidaf_combine_kernel<<<n4 / 256, 256, 0, stream>>>(context, ac, out);
}
